// AdaptiveSANet_32933809225940
// MI455X (gfx1250) — matrix operand order repaired and checked by reading, not compiled, not run on hardware
//
#include <hip/hip_runtime.h>

// ---------------------------------------------------------------------------
// AdaptiveSANet pipeline for gfx1250 (CDNA5), wave32 + WMMA bf16.
// Shapes fixed by the reference: B=4, C=512, HW=4096, HID=256.
// All heavy GEMMs arranged as NT (C[m,n] = sum_k A[m,k]*B[n,k]) so both WMMA
// operands are K-contiguous, matching the ISA fragment layouts. Wave tile is
// 32x64 (2 A-frags x 4 B-frags = 8 WMMA per 32-wide K slab). N and K are
// compile-time so addressing folds to shifts + immediate offsets. All GEMM
// memory traffic is forced to address-space(1) so the backend emits
// global_load_b128 / global_store. __launch_bounds__(256,1) lifts the
// occupancy-driven VGPR cap so the ~140-VGPR working set never spills.
// ---------------------------------------------------------------------------

typedef __attribute__((ext_vector_type(16))) __bf16 v16bf;
typedef __attribute__((ext_vector_type(8)))  __bf16 v8bf;
typedef __attribute__((ext_vector_type(8)))  float  v8f;

#define DEV __device__ __forceinline__
#define GAS __attribute__((address_space(1)))

static constexpr int Bn   = 4;
static constexpr int Cc   = 512;
static constexpr int HWn  = 4096;
static constexpr int HIDn = 256;

DEV __bf16 f2bf(float f) {
  unsigned u; __builtin_memcpy(&u, &f, 4);
  unsigned r = u + 0x7FFFu + ((u >> 16) & 1u);   // round-to-nearest-even
  unsigned short h = (unsigned short)(r >> 16);
  __bf16 o; __builtin_memcpy(&o, &h, 2);
  return o;
}

// Global-address-space accessors (force global_* instead of flat_*).
DEV v8bf  gload8(const __bf16* p) {
  return *(const GAS v8bf*)(unsigned long long)p;
}
DEV float gloadf(const float* p) {
  return *(const GAS float*)(unsigned long long)p;
}
DEV void  gstoreb(__bf16* p, __bf16 v) {
  *(GAS __bf16*)(unsigned long long)p = v;
}
DEV void  gstoref(float* p, float v) {
  *(GAS float*)(unsigned long long)p = v;
}

DEV float block_sum256(float v, float* sb) {
  int t = threadIdx.x;
  sb[t] = v; __syncthreads();
  for (int k = 128; k > 0; k >>= 1) {
    if (t < k) sb[t] += sb[t + k];
    __syncthreads();
  }
  float r = sb[0]; __syncthreads();
  return r;
}

DEV float block_max256(float v, float* sb) {
  int t = threadIdx.x;
  sb[t] = v; __syncthreads();
  for (int k = 128; k > 0; k >>= 1) {
    if (t < k) sb[t] = fmaxf(sb[t], sb[t + k]);
    __syncthreads();
  }
  float r = sb[0]; __syncthreads();
  return r;
}

// ---- per-(b,c) mean / rstd over 4096 spatial elements (mvn, ddof=1) -------
__global__ void row_stats_kernel(const float* __restrict__ x,
                                 float* __restrict__ mean,
                                 float* __restrict__ rstd) {
  __shared__ float sb[256];
  const int row = blockIdx.x;                    // b*C + c
  const float* p = x + (size_t)row * HWn;
  float s = 0.f, ss = 0.f;
  for (int i = threadIdx.x; i < HWn; i += 256) {
    float v = p[i]; s += v; ss += v * v;
  }
  float ts  = block_sum256(s, sb);
  float tss = block_sum256(ss, sb);
  if (threadIdx.x == 0) {
    float m = ts / (float)HWn;
    float var = (tss - (float)HWn * m * m) / (float)(HWn - 1);
    mean[row] = m;
    rstd[row] = rsqrtf(var + 1e-5f);
  }
}

// ---- per-(b,k) 1/L2norm over channels (coalesced along k) -----------------
__global__ void col_rnorm_kernel(const float* __restrict__ x,
                                 float* __restrict__ rn) {
  const int b = blockIdx.y;
  const int k = blockIdx.x * 256 + threadIdx.x;
  const float* p = x + (size_t)b * Cc * HWn + k;
  float s = 0.f;
  for (int c = 0; c < Cc; ++c) {
    float v = p[(size_t)c * HWn];
    s += v * v;
  }
  rn[(size_t)b * HWn + k] = 1.0f / fmaxf(sqrtf(s), 1e-12f);
}

// ---- f32 -> bf16 convert (weights) ----------------------------------------
__global__ void cvt_bf16_kernel(const float* __restrict__ in,
                                __bf16* __restrict__ out, int n) {
  int i = blockIdx.x * 256 + threadIdx.x;
  if (i < n) out[i] = f2bf(in[i]);
}

// ---- transpose [C,HW] f32 -> [HW,C] bf16 with optional normalization ------
// PM=0: (x - mean[c]) * rstd[c]   PM=1: x * rnorm[k]   PM=2: plain
template <int PM>
__global__ void pack_t_kernel(const float* __restrict__ in,
                              __bf16* __restrict__ out,
                              const float* __restrict__ p0,
                              const float* __restrict__ p1) {
  __shared__ float tile[32][33];
  const int b = blockIdx.z;
  const int k0 = blockIdx.x * 32, c0 = blockIdx.y * 32;
  const float* ip = in + (size_t)b * Cc * HWn;
  __bf16* op = out + (size_t)b * HWn * Cc;
  const int tx = threadIdx.x, ty = threadIdx.y;
#pragma unroll
  for (int i = ty; i < 32; i += 8) {
    const int c = c0 + i, k = k0 + tx;
    float v = ip[(size_t)c * HWn + k];
    if (PM == 0) v = (v - p0[b * Cc + c]) * p1[b * Cc + c];
    else if (PM == 1) v = v * p0[(size_t)b * HWn + k];
    tile[i][tx] = v;
  }
  __syncthreads();
#pragma unroll
  for (int i = ty; i < 32; i += 8) {
    const int k = k0 + i, c = c0 + tx;
    op[(size_t)k * Cc + c] = f2bf(tile[tx][i]);
  }
}

DEV v16bf cat16(v8bf lo, v8bf hi) {
  return __builtin_shufflevector(lo, hi,
           0,1,2,3,4,5,6,7,8,9,10,11,12,13,14,15);
}

// ---- WMMA NT GEMM: C[m,n] = sum_k A[m,k] * B[n,k] -------------------------
// 256 threads = 8 waves (2x4), block tile 64(M) x 256(N), wave tile 32x64.
// NC, KC compile-time (powers of two): addressing folds to shifts + imm.
// OM: 0 = bf16 store, 1 = f32 store, 2 = f32 + leaky(0.2), 3 = f32 store.
// BNE: add bias_n[n]; BME: add bias_m[m]; RSE: add resid (same layout as out).
template <int OM, bool BNE, bool BME, bool RSE, int NC, int KC>
__global__ void __launch_bounds__(256, 1)
gemm_nt_kernel(const __bf16* __restrict__ A, size_t sA,
               const __bf16* __restrict__ Bm, size_t sB,
               void* __restrict__ Out, size_t sO,
               const float* __restrict__ bias_n,
               const float* __restrict__ bias_m,
               const float* __restrict__ resid, size_t sR) {
  const int lane = threadIdx.x & 31;
  const int wave = threadIdx.x >> 5;
  const int r  = lane & 15;
  const int hi = lane >> 4;
  const int m0 = blockIdx.y * 64  + (wave >> 2) * 32;
  const int n0 = blockIdx.x * 256 + (wave & 3) * 64;
  const int bz = blockIdx.z;
  const __bf16* Ab = A  + (size_t)bz * sA;
  const __bf16* Bb = Bm + (size_t)bz * sB;

  // Loop-invariant per-lane row pointers (K-contiguous rows).
  // A frag (16x32): lane r -> row; half 'hi' supplies K = hi*8.. and 16+hi*8..
  // B frag (32x16): lane r -> column row in [N,K]; half 'hi' -> K = hi*8.. and 16+hi*8..
  const __bf16* pa[2];
  const __bf16* pb[4];
#pragma unroll
  for (int t = 0; t < 2; ++t)
    pa[t] = Ab + (size_t)(m0 + t * 16 + r) * KC + hi * 8;
#pragma unroll
  for (int j = 0; j < 4; ++j)
    pb[j] = Bb + (size_t)(n0 + j * 16 + r) * KC + hi * 8;

  v8f acc[2][4] = {};
#pragma unroll 1
  for (int ks = 0; ks < KC / 32; ++ks) {
    v16bf af[2], bfv[4];
#pragma unroll
    for (int t = 0; t < 2; ++t)
      af[t] = cat16(gload8(pa[t]), gload8(pa[t] + 16));
#pragma unroll
    for (int j = 0; j < 4; ++j)
      bfv[j] = cat16(gload8(pb[j]), gload8(pb[j] + 16));
#pragma unroll
    for (int t = 0; t < 2; ++t) pa[t] += 32;
#pragma unroll
    for (int j = 0; j < 4; ++j) pb[j] += 32;
#pragma unroll
    for (int i = 0; i < 2; ++i)
#pragma unroll
      for (int j = 0; j < 4; ++j)
        acc[i][j] = __builtin_amdgcn_wmma_f32_16x16x32_bf16(
            false, af[i], false, bfv[j], (short)0, acc[i][j], false, false);
  }

  // Epilogue (branch-free). D layout: lane (hi,r), elem e ->
  // (m = m0 + i*16 + hi*8 + e, n = n0 + j*16 + r).
  float bnv[4];
  if constexpr (BNE) {
#pragma unroll
    for (int j = 0; j < 4; ++j) bnv[j] = gloadf(bias_n + n0 + j * 16 + r);
  }
  const size_t obase = (size_t)bz * sO;
#pragma unroll
  for (int i = 0; i < 2; ++i) {
    const int mb = m0 + i * 16 + hi * 8;
    float bmv[8];
    if constexpr (BME) {
#pragma unroll
      for (int e = 0; e < 8; ++e) bmv[e] = gloadf(bias_m + mb + e);
    }
#pragma unroll
    for (int j = 0; j < 4; ++j) {
      const int n = n0 + j * 16 + r;
#pragma unroll
      for (int e = 0; e < 8; ++e) {
        const int m = mb + e;
        float v = acc[i][j][e];
        if constexpr (BNE) v += bnv[j];
        if constexpr (BME) v += bmv[e];
        if constexpr (OM == 2) v = (v > 0.0f) ? v : 0.2f * v;
        const size_t idx = obase + (size_t)m * NC + n;
        if constexpr (RSE)
          v += gloadf(resid + (size_t)bz * sR + (size_t)m * NC + n);
        if constexpr (OM == 0) gstoreb((__bf16*)Out + idx, f2bf(v));
        else                   gstoref((float*)Out + idx, v);
      }
    }
  }
}

// ---- psi: sigmoid(hmid . W2 + b2) -> clamp value per (b,k) row ------------
__global__ void psi_kernel(const float* __restrict__ hmid,
                           const float* __restrict__ W2,
                           const float* __restrict__ b2,
                           float* __restrict__ clampv) {
  __shared__ float sb[256];
  const int row = blockIdx.x;                    // b*HW + k
  float v = hmid[(size_t)row * HIDn + threadIdx.x] * W2[threadIdx.x];
  float s = block_sum256(v, sb);
  if (threadIdx.x == 0) {
    float z = s + b2[0];
    float p = 1.0f / (1.0f + __expf(-z));
    clampv[row] = p * 0.5f + 0.4f;               // VALUE_INTERVAL, FROM_VALUE
  }
}

// ---- softmax over style positions + adaptive sigmoid gate -> bf16 ---------
// Row (4096 floats, 16 KB) cached in LDS so HBM sees exactly one read.
__global__ void softmax_gate_kernel(const float* __restrict__ S,
                                    const float* __restrict__ clampv,
                                    __bf16* __restrict__ Sg) {
  __shared__ float sb[256];
  __shared__ float rowbuf[HWn];
  const int row = blockIdx.x;                    // b*HW + k
  const float* p = S + (size_t)row * HWn;
  float mx = -3.0e38f;
  for (int i = threadIdx.x; i < HWn; i += 256) {
    float v = p[i];
    rowbuf[i] = v;
    mx = fmaxf(mx, v);
  }
  mx = block_max256(mx, sb);
  float s = 0.0f;
  for (int i = threadIdx.x; i < HWn; i += 256) {
    float e = __expf(rowbuf[i] - mx);
    rowbuf[i] = e;
    s += e;
  }
  s = block_sum256(s, sb);
  const float inv = 1.0f / s;
  const float cv = clampv[row];
  __bf16* q = Sg + (size_t)row * HWn;
  for (int i = threadIdx.x; i < HWn; i += 256) {
    float prob = rowbuf[i] * inv;
    float g = 1.0f / (1.0f + __expf(-50.0f * (prob - cv)));
    q[i] = f2bf(g);
  }
}

// ---------------------------------------------------------------------------
extern "C" void kernel_launch(void* const* d_in, const int* in_sizes, int n_in,
                              void* d_out, int out_size, void* d_ws,
                              size_t ws_size, hipStream_t stream) {
  (void)in_sizes; (void)n_in; (void)out_size; (void)ws_size;
  const float* content = (const float*)d_in[0];
  const float* style   = (const float*)d_in[1];
  const float* Wf  = (const float*)d_in[2];
  const float* bfp = (const float*)d_in[3];
  const float* Wg  = (const float*)d_in[4];
  const float* bgp = (const float*)d_in[5];
  const float* Wh  = (const float*)d_in[6];
  const float* bhp = (const float*)d_in[7];
  const float* Wo  = (const float*)d_in[8];
  const float* bop = (const float*)d_in[9];
  const float* W1  = (const float*)d_in[10];
  const float* b1p = (const float*)d_in[11];
  const float* W2  = (const float*)d_in[12];
  const float* b2p = (const float*)d_in[13];
  float* out = (float*)d_out;

  char* w = (char*)d_ws;
  size_t off = 0;
  auto alloc = [&](size_t bytes) -> void* {
    void* p = w + off;
    off += (bytes + 255) & ~(size_t)255;
    return p;
  };

  const size_t BC   = (size_t)Bn * Cc;
  const size_t BHW  = (size_t)Bn * HWn;
  const size_t BCHW = (size_t)Bn * Cc * HWn;
  const size_t BHH  = (size_t)Bn * HWn * HWn;

  float* meanC  = (float*)alloc(BC * 4);
  float* rstdC  = (float*)alloc(BC * 4);
  float* meanS  = (float*)alloc(BC * 4);
  float* rstdS  = (float*)alloc(BC * 4);
  float* rnC    = (float*)alloc(BHW * 4);
  float* rnS    = (float*)alloc(BHW * 4);
  float* clampv = (float*)alloc(BHW * 4);
  __bf16* Wf_b = (__bf16*)alloc((size_t)Cc * Cc * 2);
  __bf16* Wg_b = (__bf16*)alloc((size_t)Cc * Cc * 2);
  __bf16* Wh_b = (__bf16*)alloc((size_t)Cc * Cc * 2);
  __bf16* Wo_b = (__bf16*)alloc((size_t)Cc * Cc * 2);
  __bf16* W1_b = (__bf16*)alloc((size_t)HIDn * HWn * 2);
  __bf16* cnT  = (__bf16*)alloc(BCHW * 2);   // mvn(content)^T   [HW,C]
  __bf16* snT  = (__bf16*)alloc(BCHW * 2);   // mvn(style)^T
  __bf16* styT = (__bf16*)alloc(BCHW * 2);   // style^T (Hv B-operand)
  __bf16* cfnT = (__bf16*)alloc(BCHW * 2);   // content / colnorm, ^T
  __bf16* sfnT = (__bf16*)alloc(BCHW * 2);   // style / colnorm, ^T
  __bf16* FqT  = (__bf16*)alloc(BCHW * 2);   // [HW,C]
  __bf16* GkT  = (__bf16*)alloc(BCHW * 2);   // [HW,C]
  __bf16* Hv   = (__bf16*)alloc(BCHW * 2);   // [C,HW]
  __bf16* OT   = (__bf16*)alloc(BCHW * 2);   // [HW,C]
  float*  hmid = (float*)alloc((size_t)Bn * HWn * HIDn * 4);
  __bf16* Aaff = (__bf16*)alloc(BHH * 2);    // affinity; reused later as Sg
  float*  Sraw = (float*)alloc(BHH * 4);     // raw attention scores

  // --- stage 1: statistics ---
  row_stats_kernel<<<Bn * Cc, 256, 0, stream>>>(content, meanC, rstdC);
  row_stats_kernel<<<Bn * Cc, 256, 0, stream>>>(style,   meanS, rstdS);
  col_rnorm_kernel<<<dim3(HWn / 256, Bn), 256, 0, stream>>>(content, rnC);
  col_rnorm_kernel<<<dim3(HWn / 256, Bn), 256, 0, stream>>>(style,   rnS);

  // --- stage 2: weight quantization ---
  cvt_bf16_kernel<<<(Cc * Cc) / 256, 256, 0, stream>>>(Wf, Wf_b, Cc * Cc);
  cvt_bf16_kernel<<<(Cc * Cc) / 256, 256, 0, stream>>>(Wg, Wg_b, Cc * Cc);
  cvt_bf16_kernel<<<(Cc * Cc) / 256, 256, 0, stream>>>(Wh, Wh_b, Cc * Cc);
  cvt_bf16_kernel<<<(Cc * Cc) / 256, 256, 0, stream>>>(Wo, Wo_b, Cc * Cc);
  cvt_bf16_kernel<<<(HIDn * HWn) / 256, 256, 0, stream>>>(W1, W1_b, HIDn * HWn);

  // --- stage 3: transpose + normalize packs ---
  dim3 pb(32, 8), pg(HWn / 32, Cc / 32, Bn);
  pack_t_kernel<0><<<pg, pb, 0, stream>>>(content, cnT,  meanC, rstdC);
  pack_t_kernel<0><<<pg, pb, 0, stream>>>(style,   snT,  meanS, rstdS);
  pack_t_kernel<2><<<pg, pb, 0, stream>>>(style,   styT, nullptr, nullptr);
  pack_t_kernel<1><<<pg, pb, 0, stream>>>(content, cfnT, rnC, nullptr);
  pack_t_kernel<1><<<pg, pb, 0, stream>>>(style,   sfnT, rnS, nullptr);

  const size_t sKC = (size_t)HWn * Cc;   // [HW,C] batch stride
  const size_t sCK = (size_t)Cc * HWn;   // [C,HW] batch stride
  const size_t sKK = (size_t)HWn * HWn;  // [HW,HW] batch stride

  // --- stage 4: 1x1 convs as WMMA GEMMs ---
  // FqT[k,o] = sum_c cnT[k,c] * Wf[o,c] + bf[o]    (M=HW, N=C, K=C)
  gemm_nt_kernel<0, true, false, false, Cc, Cc>
      <<<dim3(Cc / 256, HWn / 64, Bn), 256, 0, stream>>>(
      cnT, sKC, Wf_b, 0, FqT, sKC, bfp, nullptr, nullptr, 0);
  gemm_nt_kernel<0, true, false, false, Cc, Cc>
      <<<dim3(Cc / 256, HWn / 64, Bn), 256, 0, stream>>>(
      snT, sKC, Wg_b, 0, GkT, sKC, bgp, nullptr, nullptr, 0);
  // Hv[o,l] = sum_c Wh[o,c] * styT[l,c] + bh[o]    (M=C, N=HW, K=C)
  gemm_nt_kernel<0, false, true, false, HWn, Cc>
      <<<dim3(HWn / 256, Cc / 64, Bn), 256, 0, stream>>>(
      Wh_b, 0, styT, sKC, Hv, sCK, nullptr, bhp, nullptr, 0);

  // --- stage 5: affinity and attention scores (M=HW, N=HW, K=C) ---
  gemm_nt_kernel<0, false, false, false, HWn, Cc>
      <<<dim3(HWn / 256, HWn / 64, Bn), 256, 0, stream>>>(
      cfnT, sKC, sfnT, sKC, Aaff, sKK, nullptr, nullptr, nullptr, 0);
  gemm_nt_kernel<1, false, false, false, HWn, Cc>
      <<<dim3(HWn / 256, HWn / 64, Bn), 256, 0, stream>>>(
      FqT, sKC, GkT, sKC, Sraw, sKK, nullptr, nullptr, nullptr, 0);

  // --- stage 6: AEA module ---
  // hmid[k,o] = leaky(sum_l Aaff[k,l] * W1[o,l] + b1[o], 0.2)  (N=HID, K=HW)
  gemm_nt_kernel<2, true, false, false, HIDn, HWn>
      <<<dim3(HIDn / 256, HWn / 64, Bn), 256, 0, stream>>>(
      Aaff, sKK, W1_b, 0, hmid, (size_t)HWn * HIDn, b1p, nullptr, nullptr, 0);
  psi_kernel<<<Bn * HWn, 256, 0, stream>>>(hmid, W2, b2p, clampv);

  // --- stage 7: softmax + gate (Sg reuses the Aaff buffer) ---
  softmax_gate_kernel<<<Bn * HWn, 256, 0, stream>>>(Sraw, clampv, Aaff);

  // --- stage 8: O^T[k,c] = sum_l Sg[k,l] * Hv[c,l]  (M=HW, N=C, K=HW) ---
  gemm_nt_kernel<0, false, false, false, Cc, HWn>
      <<<dim3(Cc / 256, HWn / 64, Bn), 256, 0, stream>>>(
      Aaff, sKK, Hv, sCK, OT, sKC, nullptr, nullptr, nullptr, 0);

  // --- stage 9: out[o,k] = sum_c Wout[o,c]*OT[k,c] + bout[o] + content ---
  gemm_nt_kernel<3, false, true, true, HWn, Cc>
      <<<dim3(HWn / 256, Cc / 64, Bn), 256, 0, stream>>>(
      Wo_b, 0, OT, sKC, out, sCK, nullptr, bop, content, sCK);
}
